// OldRouterModel_70291434766425
// MI455X (gfx1250) — matrix operand order repaired and checked by reading, not compiled, not run on hardware
//
#include <hip/hip_runtime.h>

// ---------------- problem sizes ----------------
#define NN 10000      // nodes
#define EE 160000     // edges
#define DN 256        // node feature dim
#define DE 512        // edge feature dim
#define HH 8          // heads
#define NC 10         // classes
#define KN 5000       // top-k nodes
#define KE 80000      // top-k edges

typedef unsigned short ushort;
typedef unsigned int uint;
typedef __attribute__((ext_vector_type(8)))  __bf16 v8bf;
typedef __attribute__((ext_vector_type(16))) __bf16 v16bf;
typedef __attribute__((ext_vector_type(8)))  float  v8f;

__device__ __forceinline__ ushort f2bf(float f) {
  uint u = __float_as_uint(f);
  uint r = u + 0x7FFFu + ((u >> 16) & 1u);   // RNE
  return (ushort)(r >> 16);
}
__device__ __forceinline__ float bf2f(ushort h) {
  return __uint_as_float(((uint)h) << 16);
}
// monotone float<->uint order transform (radix select / atomic max)
__device__ __forceinline__ uint ordf(float f) {
  uint u = __float_as_uint(f);
  return (u & 0x80000000u) ? ~u : (u | 0x80000000u);
}
__device__ __forceinline__ float unord(uint u) {
  return __uint_as_float((u & 0x80000000u) ? (u & 0x7FFFFFFFu) : ~u);
}

// ---- CDNA5 async global->LDS copy (ASYNCcnt path). LDS byte offset is the
// low 32 bits of a generic pointer to __shared__ (aperture lives in [63:32]).
#define USE_ASYNC_LDS 1
__device__ __forceinline__ void cp_async_b128(void* lds, const void* gsrc) {
#if USE_ASYNC_LDS
  uint l = (uint)(uintptr_t)lds;
  asm volatile("global_load_async_to_lds_b128 %0, %1, off"
               :: "v"(l), "v"(gsrc) : "memory");
#else
  *(v8bf*)lds = *(const v8bf*)gsrc;
#endif
}
__device__ __forceinline__ void cp_async_join() {
#if USE_ASYNC_LDS
  asm volatile("s_wait_asynccnt 0x0" ::: "memory");
#endif
}

__device__ __forceinline__ v16bf cat16(v8bf lo, v8bf hi) {
  return __builtin_shufflevector(lo, hi, 0,1,2,3,4,5,6,7,8,9,10,11,12,13,14,15);
}

// ---------------- weight transpose + bf16 convert ----------------
__global__ void wtrans_kernel(const float* __restrict__ W, ushort* __restrict__ Wt,
                              int K, int Ncols, int NPad) {
  int i = blockIdx.x * 256 + threadIdx.x;
  if (i >= K * NPad) return;
  int n = i / K, kk = i - n * K;
  float v = (n < Ncols) ? W[(size_t)kk * Ncols + n] : 0.0f;
  Wt[(size_t)n * K + kk] = f2bf(v);
}

// ---------------- node score + bf16 copy ----------------
__global__ void node_score_kernel(const float* __restrict__ nf, const float* __restrict__ nw,
                                  const float* __restrict__ nb, ushort* __restrict__ nodeBf,
                                  float* __restrict__ nscore) {
  int n = blockIdx.x, d = threadIdx.x;              // 256 threads
  float v = nf[(size_t)n * DN + d];
  nodeBf[(size_t)n * DN + d] = f2bf(v);
  __shared__ float red[256];
  red[d] = v * nw[d];
  __syncthreads();
  for (int o = 128; o > 0; o >>= 1) { if (d < o) red[d] += red[d + o]; __syncthreads(); }
  if (d == 0) nscore[n] = red[0] + nb[0];
}

// ---------------- edge score + bf16 copy (fused) ----------------
__global__ void edge_score_kernel(const float* __restrict__ ef, const float* __restrict__ ew,
                                  const float* __restrict__ eb, ushort* __restrict__ ebf,
                                  float* __restrict__ escore) {
  int e = blockIdx.x, d = threadIdx.x;              // 256 threads, 2 elems each
  const float* row = ef + (size_t)e * DE;
  float v0 = row[d], v1 = row[d + 256];
  ushort* orow = ebf + (size_t)e * DE;
  orow[d] = f2bf(v0);
  orow[d + 256] = f2bf(v1);
  __shared__ float red[256];
  red[d] = v0 * ew[d] + v1 * ew[d + 256];
  __syncthreads();
  for (int o = 128; o > 0; o >>= 1) { if (d < o) red[d] += red[d + o]; __syncthreads(); }
  if (d == 0) escore[e] = red[0] + eb[0];
}

// ---------------- exact k-th largest via 4-pass radix select ----------------
struct SelState { uint prefix; uint krem; uint hist[256]; };

__global__ void sel_init_kernel(SelState* st, uint k) {
  if (threadIdx.x == 0) { st->prefix = 0u; st->krem = k; }
  st->hist[threadIdx.x] = 0u;
}
__global__ void sel_hist_kernel(const float* __restrict__ x, int n, int pass, SelState* st) {
  int i = blockIdx.x * 256 + threadIdx.x;
  if (i >= n) return;
  uint u = ordf(x[i]);
  uint mask = (pass == 3) ? 0u : (0xFFFFFFFFu << ((pass + 1) * 8));
  if ((u & mask) == (st->prefix & mask))
    atomicAdd(&st->hist[(u >> (pass * 8)) & 0xFFu], 1u);
}
__global__ void sel_step_kernel(SelState* st, int pass, float* thr_out) {
  if (threadIdx.x == 0) {
    uint krem = st->krem, cum = 0; int bin = 0;
    for (int b = 255; b >= 0; b--) {
      uint c = st->hist[b];
      if (cum + c >= krem) { bin = b; st->krem = krem - cum; break; }
      cum += c;
    }
    st->prefix |= ((uint)bin) << (pass * 8);
    if (pass == 0) *thr_out = unord(st->prefix);
  }
  __syncthreads();
  st->hist[threadIdx.x] = 0u;
}

// ---------------- masks ----------------
__global__ void node_mask_kernel(const float* __restrict__ nscore, const float* __restrict__ nthr,
                                 int* __restrict__ nmask) {
  int i = blockIdx.x * 256 + threadIdx.x;
  if (i < NN) nmask[i] = (nscore[i] >= *nthr) ? 1 : 0;
}
__global__ void emask_kernel(const float* __restrict__ escore, const float* __restrict__ ethr,
                             const int* __restrict__ nmask, const int* __restrict__ src,
                             const int* __restrict__ dst, float* __restrict__ emask) {
  int e = blockIdx.x * 256 + threadIdx.x;
  if (e < EE)
    emask[e] = ((escore[e] >= *ethr) && nmask[src[e]] && nmask[dst[e]]) ? 1.0f : 0.0f;
}

// ---------------- WMMA bf16 GEMM with LDS-staged B panel ----------------
// out[M,*] = rowdiag(mask) * A[M,KD] * Bt^T   (Bt stored [n][k])
// 8 waves/block; wave strip = 16 x (NT*16). B panel async-staged in KCH chunks.
// EPI: 0 = f32 store, 1 = bf16, 2 = bf16 + masked residual, 3 = bf16 + exact GELU
template<int NT, int EPI, int KD, int KCH>
__global__ void __launch_bounds__(256)
gemm_bf16_kernel(const ushort* __restrict__ A, const ushort* __restrict__ Bt,
                 const float* __restrict__ bias, const ushort* __restrict__ resid,
                 const float* __restrict__ rowMask, const float* __restrict__ residMask,
                 const ushort* __restrict__ zrow, void* __restrict__ outp,
                 int M, int ldOut, int nValid) {
  constexpr int LR = KCH + 8;                 // LDS row stride (bank-quad friendly)
  __shared__ ushort Bs[NT * 16 * LR];
  const int tid = threadIdx.x;
  const int lane = tid & 31, wave = tid >> 5;
  const int tileM = blockIdx.x * 128 + wave * 16;
  const int colBase = blockIdx.y * (NT * 16);
  const int mRow = tileM + (lane & 15);
  const int mLoad = (mRow < M) ? mRow : (M - 1);
  const int khalf = lane >> 4;
  const float rm = rowMask ? rowMask[mLoad] : 1.0f;
  const ushort* arow = (rm == 0.0f) ? zrow : (A + (size_t)mLoad * KD);

  v8f acc[NT];
  const v8f vzero = {0.f, 0.f, 0.f, 0.f, 0.f, 0.f, 0.f, 0.f};
#pragma unroll
  for (int t = 0; t < NT; t++) acc[t] = vzero;

  for (int kc = 0; kc < KD; kc += KCH) {
    constexpr int NCH = NT * 16 * (KCH / 8);  // 16-byte chunks in panel
#pragma unroll 1
    for (int c = tid; c < NCH; c += 256) {
      int n = c / (KCH / 8);
      int k8 = c - n * (KCH / 8);
      cp_async_b128(&Bs[n * LR + k8 * 8],
                    Bt + (size_t)(colBase + n) * KD + kc + k8 * 8);
    }
    cp_async_join();
    __syncthreads();
#pragma unroll
    for (int k0 = 0; k0 < KCH; k0 += 32) {
      // A fragment (ISA 16-bit layout): lane<16 -> K{0..7,16..23}, lane>=16 -> K{8..15,24..31}
      v8bf alo = *(const v8bf*)(arow + kc + k0 + khalf * 8);
      v8bf ahi = *(const v8bf*)(arow + kc + k0 + 16 + khalf * 8);
      v16bf af = cat16(alo, ahi);
      v16bf bfr[NT];
#pragma unroll
      for (int t = 0; t < NT; t++) {
        const ushort* bs = &Bs[(t * 16 + (lane & 15)) * LR + k0 + khalf * 8];
        bfr[t] = cat16(*(const v8bf*)bs, *(const v8bf*)(bs + 16));
      }
#pragma unroll
      for (int t = 0; t < NT; t++)
        acc[t] = __builtin_amdgcn_wmma_f32_16x16x32_bf16(
            false, af, false, bfr[t], (short)0, acc[t], false, false);
    }
    __syncthreads();
  }

  float* outF = (float*)outp;
  ushort* outH = (ushort*)outp;
#pragma unroll
  for (int t = 0; t < NT; t++) {
    int n = colBase + t * 16 + (lane & 15);
    float bn = (n < nValid) ? bias[n] : 0.0f;
#pragma unroll
    for (int r = 0; r < 8; r++) {
      int row = tileM + r + khalf * 8;
      if (row < M && n < nValid) {
        float x = acc[t][r] + bn;
        if (EPI == 2)
          x += bf2f(resid[(size_t)row * ldOut + n]) * (residMask ? residMask[row] : 1.0f);
        if (EPI == 3) x = 0.5f * x * (1.0f + erff(x * 0.70710678118654752f));
        if (EPI == 0) outF[(size_t)row * ldOut + n] = x;
        else          outH[(size_t)row * ldOut + n] = f2bf(x);
      }
    }
  }
}

// ---------------- fused K+V GEMM: one A read, two outputs ----------------
template<int KD, int KCH>
__global__ void __launch_bounds__(256)
gemm_kv_kernel(const ushort* __restrict__ A,
               const ushort* __restrict__ BtK, const ushort* __restrict__ BtV,
               const float* __restrict__ biasK, const float* __restrict__ biasV,
               const float* __restrict__ rowMask, const ushort* __restrict__ zrow,
               ushort* __restrict__ outK, ushort* __restrict__ outV, int M) {
  constexpr int NT = 4;
  constexpr int LR = KCH + 8;
  __shared__ ushort BsK[NT * 16 * LR];
  __shared__ ushort BsV[NT * 16 * LR];
  const int tid = threadIdx.x;
  const int lane = tid & 31, wave = tid >> 5;
  const int tileM = blockIdx.x * 128 + wave * 16;
  const int colBase = blockIdx.y * (NT * 16);
  const int mRow = tileM + (lane & 15);
  const int mLoad = (mRow < M) ? mRow : (M - 1);
  const int khalf = lane >> 4;
  const float rm = rowMask[mLoad];
  const ushort* arow = (rm == 0.0f) ? zrow : (A + (size_t)mLoad * KD);

  v8f accK[NT], accV[NT];
  const v8f vzero = {0.f, 0.f, 0.f, 0.f, 0.f, 0.f, 0.f, 0.f};
#pragma unroll
  for (int t = 0; t < NT; t++) { accK[t] = vzero; accV[t] = vzero; }

  for (int kc = 0; kc < KD; kc += KCH) {
    constexpr int NCH = NT * 16 * (KCH / 8);
#pragma unroll 1
    for (int c = tid; c < NCH; c += 256) {
      int n = c / (KCH / 8);
      int k8 = c - n * (KCH / 8);
      size_t go = (size_t)(colBase + n) * KD + kc + k8 * 8;
      cp_async_b128(&BsK[n * LR + k8 * 8], BtK + go);
      cp_async_b128(&BsV[n * LR + k8 * 8], BtV + go);
    }
    cp_async_join();
    __syncthreads();
#pragma unroll
    for (int k0 = 0; k0 < KCH; k0 += 32) {
      v8bf alo = *(const v8bf*)(arow + kc + k0 + khalf * 8);
      v8bf ahi = *(const v8bf*)(arow + kc + k0 + 16 + khalf * 8);
      v16bf af = cat16(alo, ahi);
      v16bf bfr[NT];
#pragma unroll
      for (int t = 0; t < NT; t++) {
        const ushort* bs = &BsK[(t * 16 + (lane & 15)) * LR + k0 + khalf * 8];
        bfr[t] = cat16(*(const v8bf*)bs, *(const v8bf*)(bs + 16));
      }
#pragma unroll
      for (int t = 0; t < NT; t++)
        accK[t] = __builtin_amdgcn_wmma_f32_16x16x32_bf16(
            false, af, false, bfr[t], (short)0, accK[t], false, false);
#pragma unroll
      for (int t = 0; t < NT; t++) {
        const ushort* bs = &BsV[(t * 16 + (lane & 15)) * LR + k0 + khalf * 8];
        bfr[t] = cat16(*(const v8bf*)bs, *(const v8bf*)(bs + 16));
      }
#pragma unroll
      for (int t = 0; t < NT; t++)
        accV[t] = __builtin_amdgcn_wmma_f32_16x16x32_bf16(
            false, af, false, bfr[t], (short)0, accV[t], false, false);
    }
    __syncthreads();
  }

#pragma unroll
  for (int t = 0; t < NT; t++) {
    int n = colBase + t * 16 + (lane & 15);
    float bk = biasK[n], bv = biasV[n];
#pragma unroll
    for (int r = 0; r < 8; r++) {
      int row = tileM + r + khalf * 8;
      if (row < M) {
        outK[(size_t)row * DE + n] = f2bf(accK[t][r] + bk);
        outV[(size_t)row * DE + n] = f2bf(accV[t][r] + bv);
      }
    }
  }
}

// ---------------- attention: scores[e][h] = q[src[e],h,:].k[e,h,:]/8 ----------------
__global__ void attn_score_kernel(const float* __restrict__ q, const ushort* __restrict__ kb,
                                  const int* __restrict__ src, float* __restrict__ scores) {
  int lane = threadIdx.x & 31, wave = threadIdx.x >> 5;
  int e = blockIdx.x * 8 + wave;
  if (e >= EE) return;
  int s = src[e];
  const float* qr = q + (size_t)s * DE + lane * 16;
  const ushort* kr = kb + (size_t)e * DE + lane * 16;
  float sum = 0.0f;
#pragma unroll
  for (int i = 0; i < 16; i++) sum += qr[i] * bf2f(kr[i]);
  sum += __shfl_xor(sum, 1);
  sum += __shfl_xor(sum, 2);
  if ((lane & 3) == 0) scores[(size_t)e * HH + (lane >> 2)] = sum * 0.125f;  // 1/sqrt(64)
}

__global__ void seg_max_kernel(const float* __restrict__ scores, const int* __restrict__ src,
                               uint* __restrict__ mu) {
  int i = blockIdx.x * 256 + threadIdx.x;
  if (i >= EE * HH) return;
  int e = i >> 3, h = i & 7;
  atomicMax(&mu[(size_t)src[e] * HH + h], ordf(scores[i]));
}

__global__ void den_agg_kernel(const float* __restrict__ scores, const uint* __restrict__ mu,
                               const ushort* __restrict__ vb, const int* __restrict__ src,
                               float* __restrict__ den, float* __restrict__ agg) {
  int lane = threadIdx.x & 31, wave = threadIdx.x >> 5;
  int e = blockIdx.x * 8 + wave;
  if (e >= EE) return;
  int s = src[e], h = lane >> 2;
  float m = unord(mu[(size_t)s * HH + h]);
  float ex = expf(scores[(size_t)e * HH + h] - m);
  if ((lane & 3) == 0) atomicAdd(&den[(size_t)s * HH + h], ex);
  const ushort* vr = vb + (size_t)e * DE + lane * 16;
  float* ar = agg + (size_t)s * DE + lane * 16;
#pragma unroll
  for (int i = 0; i < 16; i++) atomicAdd(&ar[i], ex * bf2f(vr[i]));
}

__global__ void agg_gather_kernel(const float* __restrict__ agg, const float* __restrict__ den,
                                  const int* __restrict__ src, ushort* __restrict__ out) {
  int e = blockIdx.x, d = threadIdx.x;
  int s = src[e];
  const float* ar = agg + (size_t)s * DE;
  const float* dr = den + (size_t)s * HH;
  ushort* orow = out + (size_t)e * DE;
  orow[d]       = f2bf(ar[d] / dr[d >> 6]);
  orow[d + 256] = f2bf(ar[d + 256] / dr[(d + 256) >> 6]);
}

// ---------------- host-side orchestration ----------------
extern "C" void kernel_launch(void* const* d_in, const int* in_sizes, int n_in,
                              void* d_out, int out_size, void* d_ws, size_t ws_size,
                              hipStream_t stream) {
  const float* nodeF = (const float*)d_in[0];
  const float* edgeF = (const float*)d_in[1];
  const int*   eidx  = (const int*)d_in[2];
  const int*   src   = eidx;
  const int*   dst   = eidx + EE;
  const float* nw = (const float*)d_in[4];
  const float* nb = (const float*)d_in[5];
  const float* ew = (const float*)d_in[6];
  const float* eb = (const float*)d_in[7];
  const float* Wq = (const float*)d_in[8];  const float* bq = (const float*)d_in[9];
  const float* Wk = (const float*)d_in[10]; const float* bk = (const float*)d_in[11];
  const float* Wv = (const float*)d_in[12]; const float* bv = (const float*)d_in[13];
  const float* Wo = (const float*)d_in[14]; const float* bo = (const float*)d_in[15];
  const float* W1 = (const float*)d_in[16]; const float* b1 = (const float*)d_in[17];
  const float* W2 = (const float*)d_in[18]; const float* b2 = (const float*)d_in[19];

  char* base = (char*)d_ws;
  size_t off = 0;
  auto alloc = [&](size_t bytes) -> char* {
    char* p = base + off;
    off += (bytes + 255) & ~(size_t)255;
    return p;
  };
  ushort* wqt = (ushort*)alloc((size_t)DE * DN * 2);   // [512][256]
  ushort* wkt = (ushort*)alloc((size_t)DE * DE * 2);
  ushort* wvt = (ushort*)alloc((size_t)DE * DE * 2);
  ushort* wot = (ushort*)alloc((size_t)DE * DE * 2);
  ushort* w1t = (ushort*)alloc((size_t)DE * DE * 2);
  ushort* w2t = (ushort*)alloc((size_t)16 * DE * 2);   // padded to 16 cols
  ushort* zrow = (ushort*)alloc((size_t)DE * 2);       // all-zero A row (mask redirect)
  ushort* nodeBf = (ushort*)alloc((size_t)NN * DN * 2);
  float* nscore = (float*)alloc((size_t)NN * 4);
  float* escore = (float*)alloc((size_t)EE * 4);
  int*   nmask  = (int*)alloc((size_t)NN * 4);
  float* emask  = (float*)alloc((size_t)EE * 4);
  SelState* selN = (SelState*)alloc(sizeof(SelState));
  SelState* selE = (SelState*)alloc(sizeof(SelState));
  float* thr = (float*)alloc(2 * 4);                   // [0]=node thr, [1]=edge thr
  float*  qf  = (float*)alloc((size_t)NN * DE * 4);
  ushort* ebf = (ushort*)alloc((size_t)EE * DE * 2);   // bf16 edge features (unmasked)
  ushort* kb  = (ushort*)alloc((size_t)EE * DE * 2);   // reused: aggsrc, then h
  ushort* vb  = (ushort*)alloc((size_t)EE * DE * 2);   // reused: edge_feats
  float* scores = (float*)alloc((size_t)EE * HH * 4);
  uint*  mu  = (uint*)alloc((size_t)NN * HH * 4);
  float* den = (float*)alloc((size_t)NN * HH * 4);
  float* agg = (float*)alloc((size_t)NN * DE * 4);
  (void)ws_size; (void)n_in; (void)in_sizes; (void)out_size;

  // 1) weights: transpose + bf16; zero row
  wtrans_kernel<<<(DN * DE + 255) / 256, 256, 0, stream>>>(Wq, wqt, DN, DE, DE);
  wtrans_kernel<<<(DE * DE + 255) / 256, 256, 0, stream>>>(Wk, wkt, DE, DE, DE);
  wtrans_kernel<<<(DE * DE + 255) / 256, 256, 0, stream>>>(Wv, wvt, DE, DE, DE);
  wtrans_kernel<<<(DE * DE + 255) / 256, 256, 0, stream>>>(Wo, wot, DE, DE, DE);
  wtrans_kernel<<<(DE * DE + 255) / 256, 256, 0, stream>>>(W1, w1t, DE, DE, DE);
  wtrans_kernel<<<(DE * 16 + 255) / 256, 256, 0, stream>>>(W2, w2t, DE, NC, 16);
  hipMemsetAsync(zrow, 0, (size_t)DE * 2, stream);

  // 2) router scores (+ bf16 conversions fused)
  node_score_kernel<<<NN, 256, 0, stream>>>(nodeF, nw, nb, nodeBf, nscore);
  edge_score_kernel<<<EE, 256, 0, stream>>>(edgeF, ew, eb, ebf, escore);

  // 3) exact top-k thresholds (4-pass radix select, fully on device)
  sel_init_kernel<<<1, 256, 0, stream>>>(selN, KN);
  sel_init_kernel<<<1, 256, 0, stream>>>(selE, KE);
  for (int pass = 3; pass >= 0; pass--) {
    sel_hist_kernel<<<(NN + 255) / 256, 256, 0, stream>>>(nscore, NN, pass, selN);
    sel_step_kernel<<<1, 256, 0, stream>>>(selN, pass, thr + 0);
    sel_hist_kernel<<<(EE + 255) / 256, 256, 0, stream>>>(escore, EE, pass, selE);
    sel_step_kernel<<<1, 256, 0, stream>>>(selE, pass, thr + 1);
  }

  // 4) masks
  node_mask_kernel<<<(NN + 255) / 256, 256, 0, stream>>>(nscore, thr + 0, nmask);
  emask_kernel<<<(EE + 255) / 256, 256, 0, stream>>>(escore, thr + 1, nmask, src, dst, emask);

  // 5) q GEMM + fused K/V GEMM (WMMA bf16, LDS-staged B)
  {
    dim3 g((NN + 127) / 128, DE / 64);
    gemm_bf16_kernel<4, 0, DN, 256><<<g, 256, 0, stream>>>(
        nodeBf, wqt, bq, nullptr, nullptr, nullptr, zrow, qf, NN, DE, DE);
  }
  {
    dim3 g(EE / 128, DE / 64);
    gemm_kv_kernel<DE, 128><<<g, 256, 0, stream>>>(
        ebf, wkt, wvt, bk, bv, emask, zrow, kb, vb, EE);
  }

  // 6) segment softmax + weighted aggregation
  hipMemsetAsync(mu, 0, (size_t)NN * HH * 4, stream);    // ordf-space minimum
  hipMemsetAsync(den, 0, (size_t)NN * HH * 4, stream);
  hipMemsetAsync(agg, 0, (size_t)NN * DE * 4, stream);
  attn_score_kernel<<<EE / 8, 256, 0, stream>>>(qf, kb, src, scores);
  seg_max_kernel<<<(EE * HH + 255) / 256, 256, 0, stream>>>(scores, src, mu);
  den_agg_kernel<<<EE / 8, 256, 0, stream>>>(scores, mu, vb, src, den, agg);

  // 7) gather agg[src]/den -> bf16 A matrix (reuse kb)
  agg_gather_kernel<<<EE, 256, 0, stream>>>(agg, den, src, kb);

  // 8) Wo GEMM with masked residual(+ef) -> edge_feats bf16 (reuse vb)
  {
    dim3 g(EE / 128, DE / 64);
    gemm_bf16_kernel<4, 2, DE, 256><<<g, 256, 0, stream>>>(
        kb, wot, bo, ebf, nullptr, emask, zrow, vb, EE, DE, DE);
  }
  // 9) W1 GEMM with exact GELU -> h bf16 (reuse kb)
  {
    dim3 g(EE / 128, DE / 64);
    gemm_bf16_kernel<4, 3, DE, 256><<<g, 256, 0, stream>>>(
        vb, w1t, b1, nullptr, nullptr, nullptr, zrow, kb, EE, DE, DE);
  }
  // 10) W2 GEMM -> d_out f32 [E,10]
  {
    dim3 g(EE / 128, 1);
    gemm_bf16_kernel<1, 0, DE, 256><<<g, 256, 0, stream>>>(
        kb, w2t, b2, nullptr, nullptr, nullptr, zrow, d_out, EE, NC, NC);
  }
}
